// GPT2Block_76132590289001
// MI455X (gfx1250) — matrix operand order repaired and checked by reading, not compiled, not run on hardware
//
#include <hip/hip_runtime.h>
#include <hip/hip_bf16.h>

typedef __attribute__((ext_vector_type(16))) __bf16 v16bf;
typedef __attribute__((ext_vector_type(8)))  float  v8f;
typedef __attribute__((ext_vector_type(8)))  unsigned int v8u;
typedef __attribute__((ext_vector_type(4)))  unsigned int v4u;
typedef __attribute__((ext_vector_type(8)))  int v8i;
typedef __attribute__((ext_vector_type(4)))  int v4i;

#define D_MODEL 256
#define D_HID   128
#define SEQ     1024
#define BATCH   32
#define NTOK    (BATCH * SEQ)

// ---------------------------------------------------------------------------
// Fragment loaders for WMMA 16x16x32 bf16, contiguous k order (k = 16*(lane/16) + s) in A and B
// ---------------------------------------------------------------------------

// B fragment (32x16, K x N). Bt is stored [N][K] row-major, so each lane reads
// 16 contiguous bf16 (32B): lane n = l%16 selects the row, lanes 16..31 take
// the K+16 half. Works for global or LDS-backed pointers.
__device__ __forceinline__ v16bf load_bfrag(const __bf16* Bt, int ldk, int n0, int k0) {
  const int lane = threadIdx.x & 31;
  const __bf16* p = Bt + (size_t)(n0 + (lane & 15)) * ldk + (k0 + ((lane >> 4) << 4));
  const uint4* q = (const uint4*)p;
  uint4 lo = q[0];
  uint4 hi = q[1];
  v8u u;
  u[0] = lo.x; u[1] = lo.y; u[2] = lo.z; u[3] = lo.w;
  u[4] = hi.x; u[5] = hi.y; u[6] = hi.z; u[7] = hi.w;
  return __builtin_bit_cast(v16bf, u);
}

// A fragment (16x32, M x K), A row-major [M][K]. Lane l: row = l%16,
// kbase = (l/16)*16; VGPR v: K = kbase+2v..+1
__device__ __forceinline__ v16bf load_afrag(const __bf16* A, int ldk, int m0, int k0) {
  const int lane = threadIdx.x & 31;
  const int r  = lane & 15;
  const int kb = (lane >> 4) << 4;
  const __bf16* row = A + (size_t)(m0 + r) * ldk + k0;
  v8u u;
#pragma unroll
  for (int v = 0; v < 8; ++v) {
    const int kk = kb + (v << 1);
    u[v] = *(const unsigned int*)(row + kk);
  }
  return __builtin_bit_cast(v16bf, u);
}

__device__ __forceinline__ v8f wmma_bf16(v16bf a, v16bf b, v8f c) {
  return __builtin_amdgcn_wmma_f32_16x16x32_bf16(false, a, false, b, (short)0, c, false, false);
}

// ---------------------------------------------------------------------------
// Tensor Data Mover: 2-D tile load Global -> LDS (CDNA5 ISA).
// Dims / strides in element (data_size=2B) units; addresses in bytes.
// This toolchain's builtin takes 6 args (g0 v4u, g1 v8i, g2 v4i, g3 v4i,
// extra v8i, cpol i32).
// ---------------------------------------------------------------------------
__device__ __forceinline__ void tdm_load_2d(unsigned lds_off, const void* gptr,
                                            unsigned tensor_d0, unsigned tensor_d1,
                                            unsigned tile_d0, unsigned tile_d1,
                                            unsigned stride0) {
  const unsigned long long ga = (unsigned long long)(size_t)gptr;
  v4u g0;
  g0[0] = 1u;                                                  // count=1 (valid), user mode
  g0[1] = lds_off;                                             // lds_addr (bytes)
  g0[2] = (unsigned)ga;                                        // global_addr[31:0]
  g0[3] = (unsigned)((ga >> 32) & 0x01FFFFFFu) | 0x80000000u;  // addr[56:32] | type=2
  v8i g1;
  g1[0] = 0x00010000;                                          // data_size=1 (2 bytes)
  g1[1] = (int)((tensor_d0 & 0xFFFFu) << 16);                  // tensor_dim0[15:0]
  g1[2] = (int)((tensor_d0 >> 16) | ((tensor_d1 & 0xFFFFu) << 16));
  g1[3] = (int)((tensor_d1 >> 16) | ((tile_d0 & 0xFFFFu) << 16));  // tile_dim0
  g1[4] = (int)(tile_d1 & 0xFFFFu);                            // tile_dim1 (tile_dim2=0)
  g1[5] = (int)stride0;                                        // tensor_dim0_stride[31:0]
  g1[6] = 0;
  g1[7] = 0;
  v4i z4 = {0, 0, 0, 0};
  v8i z8 = {0, 0, 0, 0, 0, 0, 0, 0};
  __builtin_amdgcn_tensor_load_to_lds(g0, g1, z4, z4, z8, 0);
}

#define WAIT_DS0()     asm volatile("s_wait_dscnt 0x0" ::: "memory")
#define WAIT_TENSOR0() asm volatile("s_wait_tensorcnt 0x0" ::: "memory")
#define WAIT_TENSOR2() asm volatile("s_wait_tensorcnt 0x2" ::: "memory")

// ---------------------------------------------------------------------------
// Weight transpose + f32 -> bf16 : src[K][N] -> dst[N][K]
// ---------------------------------------------------------------------------
__global__ void k_transpose(const float* __restrict__ src, __bf16* __restrict__ dst,
                            int K, int N) {
  const int idx = blockIdx.x * blockDim.x + threadIdx.x;
  if (idx >= K * N) return;
  const int n = idx / K;
  const int k = idx - n * K;
  dst[idx] = (__bf16)src[k * N + n];
}

// ---------------------------------------------------------------------------
// LayerNorm over D_MODEL=256: one wave per token, 8 floats per lane.
// ---------------------------------------------------------------------------
__global__ void k_layernorm(const float* __restrict__ x, const float* __restrict__ sc,
                            const float* __restrict__ bi, __bf16* __restrict__ out) {
  const int token = blockIdx.x * (blockDim.x >> 5) + (threadIdx.x >> 5);
  const int lane  = threadIdx.x & 31;
  const float* row = x + (size_t)token * D_MODEL;
  const float4* rv = (const float4*)row;
  float4 a = rv[lane * 2];
  float4 b = rv[lane * 2 + 1];
  float vals[8] = {a.x, a.y, a.z, a.w, b.x, b.y, b.z, b.w};
  float s = 0.f;
#pragma unroll
  for (int i = 0; i < 8; ++i) s += vals[i];
#pragma unroll
  for (int off = 1; off < 32; off <<= 1) s += __shfl_xor(s, off, 32);
  const float mu = s * (1.0f / 256.0f);
  float var = 0.f;
#pragma unroll
  for (int i = 0; i < 8; ++i) { float d = vals[i] - mu; var += d * d; }
#pragma unroll
  for (int off = 1; off < 32; off <<= 1) var += __shfl_xor(var, off, 32);
  var *= (1.0f / 256.0f);
  const float rstd = rsqrtf(var + 1e-5f);
  __bf16* orow = out + (size_t)token * D_MODEL;
#pragma unroll
  for (int i = 0; i < 8; ++i) {
    const int c = lane * 8 + i;
    orow[c] = (__bf16)((vals[i] - mu) * rstd * sc[c] + bi[c]);
  }
}

// ---------------------------------------------------------------------------
// Generic WMMA GEMM: C[M][N] = A[M][K](bf16) @ Bt[N][K](bf16) + bias
// 4 waves per block, each wave computes a 16x64 tile.
// ---------------------------------------------------------------------------
template <bool RELU, bool ADD_RES, bool OUT_BF16>
__global__ void k_gemm(const __bf16* __restrict__ A, const __bf16* __restrict__ Bt,
                       const float* __restrict__ bias, const float* __restrict__ res,
                       float* __restrict__ outF, __bf16* __restrict__ outB,
                       int N, int K) {
  const int wave = threadIdx.x >> 5;
  const int m0 = blockIdx.x * 64 + wave * 16;
  const int n0 = blockIdx.y * 64;
  const int lane = threadIdx.x & 31;
  v8f acc[4] = {};
  for (int k0 = 0; k0 < K; k0 += 32) {
    v16bf a = load_afrag(A, K, m0, k0);
#pragma unroll
    for (int t = 0; t < 4; ++t)
      acc[t] = wmma_bf16(a, load_bfrag(Bt, K, n0 + t * 16, k0), acc[t]);
  }
  const int nl = lane & 15, hl = lane >> 4;
#pragma unroll
  for (int t = 0; t < 4; ++t) {
    const int n = n0 + t * 16 + nl;
    const float bv = bias[n];
#pragma unroll
    for (int v = 0; v < 8; ++v) {
      const size_t m = (size_t)(m0 + v + hl * 8);
      float val = acc[t][v] + bv;
      if (RELU) val = fmaxf(val, 0.0f);
      if (ADD_RES) val += res[m * N + n];
      if (OUT_BF16) outB[m * N + n] = (__bf16)val;
      else          outF[m * N + n] = val;
    }
  }
}

// ---------------------------------------------------------------------------
// QKV GEMM: h1 @ wqkvT (N=768,K=256); scatter into Q, K (row-major) and
// V transposed [b][d][s] so attention's P@V B-fragments are contiguous.
// ---------------------------------------------------------------------------
__global__ void k_qkv(const __bf16* __restrict__ A, const __bf16* __restrict__ Bt,
                      const float* __restrict__ bias,
                      __bf16* __restrict__ Q, __bf16* __restrict__ Kc,
                      __bf16* __restrict__ Vt) {
  const int wave = threadIdx.x >> 5;
  const int m0 = blockIdx.x * 64 + wave * 16;
  const int n0 = blockIdx.y * 64;
  const int lane = threadIdx.x & 31;
  v8f acc[4] = {};
  for (int k0 = 0; k0 < D_MODEL; k0 += 32) {
    v16bf a = load_afrag(A, D_MODEL, m0, k0);
#pragma unroll
    for (int t = 0; t < 4; ++t)
      acc[t] = wmma_bf16(a, load_bfrag(Bt, D_MODEL, n0 + t * 16, k0), acc[t]);
  }
  const int nl = lane & 15, hl = lane >> 4;
#pragma unroll
  for (int t = 0; t < 4; ++t) {
    const int n = n0 + t * 16 + nl;
    const float bv = bias[n];
#pragma unroll
    for (int v = 0; v < 8; ++v) {
      const int m = m0 + v + hl * 8;
      const __bf16 val = (__bf16)(acc[t][v] + bv);
      if (n < 256) {
        Q[(size_t)m * D_MODEL + n] = val;
      } else if (n < 512) {
        Kc[(size_t)m * D_MODEL + (n - 256)] = val;
      } else {
        const int bb = m >> 10, ss = m & 1023;
        Vt[((size_t)bb * D_MODEL + (n - 512)) * SEQ + ss] = val;
      }
    }
  }
}

// ---------------------------------------------------------------------------
// Flash-style causal attention, TDM-fed. One wave per (batch, 16-query tile).
// Per 32-key block: K-tile (32x256, contiguous) and V-tile (256x32, stride
// 1024) are DMA'd into double-buffered LDS by the Tensor Data Mover while the
// previous block computes. Scores via 16 WMMAs (K-dim=256) from ds_load_b128
// fragments; online softmax with 16-lane shuffle reductions; P staged through
// LDS (C-layout -> A-layout); P@V via 16 WMMAs into a 16x256 f32 accumulator.
// ---------------------------------------------------------------------------
__global__ void __launch_bounds__(32)
k_attn(const __bf16* __restrict__ Q, const __bf16* __restrict__ Kmat,
       const __bf16* __restrict__ Vt, __bf16* __restrict__ O) {
  __shared__ __align__(16) __bf16 kt[2][32 * D_MODEL];   // [key][feature]
  __shared__ __align__(16) __bf16 vt[2][D_MODEL * 32];   // [feature][key]
  __shared__ __align__(16) __bf16 pbuf[16 * 32];
  const int b  = blockIdx.y;
  const int q0 = blockIdx.x * 16;
  const int lane = threadIdx.x & 31;
  const int nl = lane & 15, hl = lane >> 4;
  const __bf16* Qb = Q    + (size_t)b * SEQ * D_MODEL;
  const __bf16* Kb = Kmat + (size_t)b * SEQ * D_MODEL;
  const __bf16* Vb = Vt   + (size_t)b * D_MODEL * SEQ;

  v16bf qf[8];
#pragma unroll
  for (int kc = 0; kc < 8; ++kc) qf[kc] = load_afrag(Qb, D_MODEL, q0, kc * 32);

  v8f o[16] = {};
  float mrow[8], lrow[8];
#pragma unroll
  for (int v = 0; v < 8; ++v) { mrow[v] = -3.0e38f; lrow[v] = 0.0f; }

  const int nblk = (q0 + 47) >> 5;  // key blocks needed for causal coverage

  // Prefetch key block 0 into buffer 0 (K tile: 1-D contiguous 8192 elems;
  // V tile: 2-D 32-wide x 256 rows with stride SEQ).
  tdm_load_2d((unsigned)(size_t)&kt[0][0], Kb, 8192u, 1u, 8192u, 1u, 8192u);
  tdm_load_2d((unsigned)(size_t)&vt[0][0], Vb, SEQ, D_MODEL, 32u, D_MODEL, SEQ);

  for (int kb = 0; kb < nblk; ++kb) {
    const int k0 = kb << 5;
    const int cur = kb & 1;
    if (kb + 1 < nblk) {
      const int nx = cur ^ 1;
      WAIT_DS0();  // prior reads of buffer nx complete before TDM overwrites it
      tdm_load_2d((unsigned)(size_t)&kt[nx][0], Kb + (size_t)(k0 + 32) * D_MODEL,
                  8192u, 1u, 8192u, 1u, 8192u);
      tdm_load_2d((unsigned)(size_t)&vt[nx][0], Vb + (k0 + 32),
                  SEQ, D_MODEL, 32u, D_MODEL, SEQ);
      // 2 TDMs for the *next* block may remain in flight; TDMs complete
      // in-order, so TENSORcnt <= 2 guarantees this block's tiles landed.
      WAIT_TENSOR2();
    } else {
      WAIT_TENSOR0();
    }

    const __bf16* kcur = &kt[cur][0];
    const __bf16* vcur = &vt[cur][0];

    v8f s0 = {}, s1 = {};
#pragma unroll
    for (int kc = 0; kc < 8; ++kc) {
      s0 = wmma_bf16(qf[kc], load_bfrag(kcur, D_MODEL, 0,  kc * 32), s0);
      s1 = wmma_bf16(qf[kc], load_bfrag(kcur, D_MODEL, 16, kc * 32), s1);
    }
    float alpha[8];
#pragma unroll
    for (int v = 0; v < 8; ++v) {
      const int qi = q0 + v + hl * 8;
      const float a = (k0 + nl      <= qi) ? s0[v] * 0.0625f : -10000.0f;
      const float c = (k0 + 16 + nl <= qi) ? s1[v] * 0.0625f : -10000.0f;
      float mx = fmaxf(a, c);
#pragma unroll
      for (int off = 1; off < 16; off <<= 1) mx = fmaxf(mx, __shfl_xor(mx, off, 32));
      const float nm = fmaxf(mrow[v], mx);
      alpha[v] = __expf(mrow[v] - nm);
      const float p0 = __expf(a - nm);
      const float p1 = __expf(c - nm);
      float rs = p0 + p1;
#pragma unroll
      for (int off = 1; off < 16; off <<= 1) rs += __shfl_xor(rs, off, 32);
      lrow[v] = lrow[v] * alpha[v] + rs;
      mrow[v] = nm;
      pbuf[(v + hl * 8) * 32 + nl]      = (__bf16)p0;
      pbuf[(v + hl * 8) * 32 + 16 + nl] = (__bf16)p1;
    }
#pragma unroll
    for (int t = 0; t < 16; ++t)
#pragma unroll
      for (int v = 0; v < 8; ++v) o[t][v] *= alpha[v];
    __syncthreads();
    const v16bf pf = load_afrag(pbuf, 32, 0, 0);
    __syncthreads();
#pragma unroll
    for (int t = 0; t < 16; ++t)
      o[t] = wmma_bf16(pf, load_bfrag(vcur, 32, t * 16, 0), o[t]);
  }

#pragma unroll
  for (int v = 0; v < 8; ++v) {
    const float inv = 1.0f / lrow[v];
    const size_t row = (size_t)b * SEQ + q0 + v + hl * 8;
#pragma unroll
    for (int t = 0; t < 16; ++t)
      O[row * D_MODEL + t * 16 + nl] = (__bf16)(o[t][v] * inv);
  }
}

// ---------------------------------------------------------------------------
// Launch
// ---------------------------------------------------------------------------
extern "C" void kernel_launch(void* const* d_in, const int* in_sizes, int n_in,
                              void* d_out, int out_size, void* d_ws, size_t ws_size,
                              hipStream_t stream) {
  (void)in_sizes; (void)n_in; (void)out_size; (void)ws_size;
  const float* x      = (const float*)d_in[0];
  const float* ln1_s  = (const float*)d_in[1];
  const float* ln1_b  = (const float*)d_in[2];
  const float* w_qkv  = (const float*)d_in[3];
  const float* b_qkv  = (const float*)d_in[4];
  const float* w_proj = (const float*)d_in[5];
  const float* b_proj = (const float*)d_in[6];
  const float* ln2_s  = (const float*)d_in[7];
  const float* ln2_b  = (const float*)d_in[8];
  const float* w_fc1  = (const float*)d_in[9];
  const float* b_fc1  = (const float*)d_in[10];
  const float* w_fc2  = (const float*)d_in[11];
  const float* b_fc2  = (const float*)d_in[12];
  float* out = (float*)d_out;

  char* ws = (char*)d_ws;
  size_t off = 0;
  auto alloc = [&](size_t bytes) -> void* {
    void* p = ws + off;
    off = (off + bytes + 255) & ~(size_t)255;
    return p;
  };
  __bf16* wqkvT  = (__bf16*)alloc((size_t)768 * 256 * 2);
  __bf16* wprojT = (__bf16*)alloc((size_t)256 * 256 * 2);
  __bf16* wfc1T  = (__bf16*)alloc((size_t)128 * 256 * 2);
  __bf16* wfc2T  = (__bf16*)alloc((size_t)256 * 128 * 2);
  __bf16* h1     = (__bf16*)alloc((size_t)NTOK * D_MODEL * 2);  // LN1 out; reused as attn O
  __bf16* Qb     = (__bf16*)alloc((size_t)NTOK * D_MODEL * 2);
  __bf16* Kb     = (__bf16*)alloc((size_t)NTOK * D_MODEL * 2);  // reused as h2 (LN2 out)
  __bf16* Vt     = (__bf16*)alloc((size_t)NTOK * D_MODEL * 2);
  float*  x2     = (float*) alloc((size_t)NTOK * D_MODEL * 4);
  __bf16* h3     = (__bf16*)alloc((size_t)NTOK * D_HID * 2);
  __bf16* Ob = h1;
  __bf16* h2 = Kb;

  // Weight prep (bf16, [N][K])
  k_transpose<<<(768 * 256 + 255) / 256, 256, 0, stream>>>(w_qkv, wqkvT, 256, 768);
  k_transpose<<<(256 * 256 + 255) / 256, 256, 0, stream>>>(w_proj, wprojT, 256, 256);
  k_transpose<<<(256 * 128 + 255) / 256, 256, 0, stream>>>(w_fc1, wfc1T, 256, 128);
  k_transpose<<<(128 * 256 + 255) / 256, 256, 0, stream>>>(w_fc2, wfc2T, 128, 256);

  // LN1
  k_layernorm<<<NTOK / 8, 256, 0, stream>>>(x, ln1_s, ln1_b, h1);
  // QKV projection
  k_qkv<<<dim3(NTOK / 64, 768 / 64), 128, 0, stream>>>(h1, wqkvT, b_qkv, Qb, Kb, Vt);
  // Causal attention (TDM-fed)
  k_attn<<<dim3(SEQ / 16, BATCH), 32, 0, stream>>>(Qb, Kb, Vt, Ob);
  // Output projection + residual
  k_gemm<false, true, false><<<dim3(NTOK / 64, 256 / 64), 128, 0, stream>>>(
      Ob, wprojT, b_proj, x, x2, nullptr, 256, 256);
  // LN2
  k_layernorm<<<NTOK / 8, 256, 0, stream>>>(x2, ln2_s, ln2_b, h2);
  // FC1 + ReLU
  k_gemm<true, false, true><<<dim3(NTOK / 64, 128 / 64), 128, 0, stream>>>(
      h2, wfc1T, b_fc1, nullptr, nullptr, h3, 128, 256);
  // FC2 + residual -> out
  k_gemm<false, true, false><<<dim3(NTOK / 64, 256 / 64), 128, 0, stream>>>(
      h3, wfc2T, b_fc2, x2, out, nullptr, 256, 128);
}
